// DySample_56221121904703
// MI455X (gfx1250) — compile-verified, matrix operand order checked by reading, not run on hardware
//
#include <hip/hip_runtime.h>
#include <hip/hip_bf16.h>

// DySample fused implementation for gfx1250 (MI455X), wave32 + WMMA f16.
//
// Kernel 1: offset/scope 1x1 conv (WMMA GEMM, M=16 px, N=64, K=64)
//           + sigmoid modulation + init_pos -> sampling grid in ws.
// Kernel 2: bilinear grid-sample of x (37.7MB, L2-resident) fused with
//           end 1x1 conv (WMMA GEMM, 64 px x 64 oc tile/block) + bias.
//
// LDS tiles are K-contiguous ([m][k] / [n][k], row stride 72 halves = 144B,
// 16B-aligned) so each WMMA fragment is two ds_load_b128 (the ISA 16-bit
// A/B layout is two contiguous 8-half runs per lane).

#define BB 4
#define CC 64
#define HH 192
#define WW 192
#define SH 384
#define SW 384

#define XS 72   // A-tile row stride in halves
#define NS 72   // B-tile row stride in halves
#define DSO 68  // D-stage row stride in floats (bank-conflict padding)

typedef __attribute__((ext_vector_type(16))) _Float16 v16h;
typedef __attribute__((ext_vector_type(8)))  float    v8f;

union FragH { v16h v; _Float16 e[16]; uint4 q[2]; };
union V8F   { v8f  v; float    e[8]; };
union H8    { _Float16 e[8]; uint4 q; };

// Build a 16x32 f16 A/B fragment from a K-contiguous LDS row.
__device__ __forceinline__ v16h load_frag(const _Float16* row, int kb, int lane) {
    const uint4* p = (const uint4*)(row + kb * 32 + ((lane >> 4) << 3));
    FragH f;
    f.q[0] = p[0];   // K = base .. base+7
    f.q[1] = p[2];   // K = base+16 .. base+23
    return f.v;
}

__device__ __forceinline__ float fast_sigmoid(float v) {
    return __builtin_amdgcn_rcpf(1.0f + __expf(-v));   // v_rcp_f32, no div seq
}

// ---------------------------------------------------------------------------
// Kernel 1: offsets -> sampling grid
// ---------------------------------------------------------------------------
__global__ __launch_bounds__(128) void dysample_offset_kernel(
    const float* __restrict__ x,         // [B][C][H][W]
    const float* __restrict__ offset_w,  // [32][C]
    const float* __restrict__ offset_b,  // [32]
    const float* __restrict__ scope_w,   // [32][C]
    float* __restrict__ grid)            // [B][4][SH][SW][2] (ws)
{
    __shared__ __align__(16) _Float16 xlds[16 * XS];  // A [m=px][k=c]
    __shared__ __align__(16) _Float16 wlds[64 * NS];  // B [n][k=c] (n<32 off)
    __shared__ float dlds[16 * DSO];                  // D [px][n]

    const int tid  = threadIdx.x;
    const int lane = tid & 31;
    const int wave = tid >> 5;

    int blk = blockIdx.x;
    const int wt = blk % (WW / 16); blk /= (WW / 16);
    const int h  = blk % HH;        blk /= HH;
    const int b  = blk;
    const int w0 = wt * 16;

    // stage x tile: pixel p, 8 channels -> one contiguous b128 LDS store
    {
        const int p = tid & 15, cb = tid >> 4;
        const float* xp = x + (((size_t)b * CC) * HH + h) * WW + w0 + p;
        H8 t;
#pragma unroll
        for (int i = 0; i < 8; ++i)
            t.e[i] = (_Float16)xp[(size_t)(cb * 8 + i) * (HH * WW)];
        *(uint4*)&xlds[p * XS + cb * 8] = t.q;
    }
    // stage weights [n][c]: row-major contiguous copy (32 f32 per thread)
    {
        const int n = tid >> 1, half = tid & 1;
        const float* src = (n < 32) ? (offset_w + n * CC) : (scope_w + (n - 32) * CC);
        H8 t;
#pragma unroll
        for (int j = 0; j < 4; ++j) {
#pragma unroll
            for (int i = 0; i < 8; ++i)
                t.e[i] = (_Float16)src[half * 32 + j * 8 + i];
            *(uint4*)&wlds[n * NS + half * 32 + j * 8] = t.q;
        }
    }
    __syncthreads();

    const _Float16* arow = xlds + (lane & 15) * XS;
    const _Float16* brow = wlds + ((lane & 15) + wave * 16) * NS;

    v8f acc = {};
#pragma unroll
    for (int kb = 0; kb < 2; ++kb) {
        v16h a  = load_frag(arow, kb, lane);
        v16h bm = load_frag(brow, kb, lane);
        acc = __builtin_amdgcn_wmma_f32_16x16x32_f16(false, a, false, bm,
                                                     (short)0, acc, false, false);
    }
    {   // D: vgpr r -> M = r + 8*(lane>=16), N = lane%16 (+wave*16)
        V8F d; d.v = acc;
        const int n = (lane & 15) + wave * 16;
        const int pb = (lane >> 4) << 3;
#pragma unroll
        for (int r = 0; r < 8; ++r)
            dlds[(pb + r) * DSO + n] = d.e[r];
    }
    __syncthreads();

    // epilogue: pixel-space sample coords gx = w + off_x, clamped to border.
    {
        const int p  = tid & 15;
        const int j  = tid >> 4;   // (group, s1)
        const int gg = j >> 1;
        const int s1 = j & 1;
        const int w  = w0 + p;
        const int Y  = 2 * h + s1;
#pragma unroll
        for (int s2 = 0; s2 < 2; ++s2) {
            int o  = gg * 4 + s1 * 2 + s2;  // x-offset channel (d=0)
            int oy = 16 + o;                // y-offset channel (d=1)
            float ox  = dlds[p * DSO + o]  + offset_b[o];
            float oyv = dlds[p * DSO + oy] + offset_b[oy];
            float sx = fast_sigmoid(dlds[p * DSO + 32 + o]);
            float sy = fast_sigmoid(dlds[p * DSO + 48 + o]);
            float offx = ox  * sx * 0.5f + ((float)s2 - 0.5f) * 0.5f;
            float offy = oyv * sy * 0.5f + ((float)s1 - 0.5f) * 0.5f;
            float gx = fminf(fmaxf((float)w + offx, 0.0f), (float)(WW - 1));
            float gy = fminf(fmaxf((float)h + offy, 0.0f), (float)(HH - 1));
            int X = 2 * w + s2;
            size_t gi = ((((size_t)b * 4 + gg) * SH + Y) * SW + X) * 2;
            grid[gi + 0] = gx;
            grid[gi + 1] = gy;
        }
    }
}

// ---------------------------------------------------------------------------
// Kernel 2: bilinear grid-sample fused with end 1x1 conv + bias.
// Tile: 64 px x 64 oc per 128-thread block; wave w owns pixels [16w,16w+16)
// and sweeps all 4 N-tiles with 4 accumulators (8 WMMAs/wave).
// ---------------------------------------------------------------------------
__global__ __launch_bounds__(128) void dysample_sample_gemm_kernel(
    const float* __restrict__ x,      // [B][C][H][W]
    const float* __restrict__ grid,   // [B][4][SH][SW][2]
    const float* __restrict__ end_w,  // [C][C]
    const float* __restrict__ end_b,  // [C]
    float* __restrict__ out)          // [B][C][SH][SW]
{
    __shared__ __align__(16) _Float16 slds[64 * XS];  // sampled A [m=px][k=c]
    __shared__ __align__(16) _Float16 wlds[64 * NS];  // B [n=o][k=c]

    const int tid  = threadIdx.x;
    const int lane = tid & 31;
    const int wave = tid >> 5;

    int blk = blockIdx.x;
    const int xt = blk % (SW / 64); blk /= (SW / 64);
    const int Y  = blk % SH;        blk /= SH;
    const int b  = blk;
    const int X0 = xt * 64;

    // stage end_w [o][c]: row-major contiguous copy
    {
        const int o = tid >> 1, half = tid & 1;
        const float* src = end_w + o * CC;
        H8 t;
#pragma unroll
        for (int j = 0; j < 4; ++j) {
#pragma unroll
            for (int i = 0; i < 8; ++i)
                t.e[i] = (_Float16)src[half * 32 + j * 8 + i];
            *(uint4*)&wlds[o * NS + half * 32 + j * 8] = t.q;
        }
    }

    // bilinear sample: thread -> 4 pixels x 8 channels of one group
    {
        const int p0 = tid & 15;
        const int cb = tid >> 4;
        const int gg = cb >> 1;          // channel c uses grid of group c/16
#pragma unroll
        for (int q = 0; q < 4; ++q) {
            const int p = p0 + q * 16;
            const int X = X0 + p;
            size_t gi = ((((size_t)b * 4 + gg) * SH + Y) * SW + X) * 2;
            float gx = grid[gi], gy = grid[gi + 1];
            float x0f = floorf(gx), y0f = floorf(gy);
            float wx = gx - x0f, wy = gy - y0f;
            int ix0 = (int)x0f, iy0 = (int)y0f;
            int ix1 = min(ix0 + 1, WW - 1), iy1 = min(iy0 + 1, HH - 1);
            float w00 = (1.0f - wx) * (1.0f - wy), w01 = wx * (1.0f - wy);
            float w10 = (1.0f - wx) * wy,          w11 = wx * wy;
            H8 t;
#pragma unroll
            for (int i = 0; i < 8; ++i) {
                int c = cb * 8 + i;
                const float* xp = x + ((size_t)b * CC + c) * (HH * WW);
                float v = xp[iy0 * WW + ix0] * w00 + xp[iy0 * WW + ix1] * w01 +
                          xp[iy1 * WW + ix0] * w10 + xp[iy1 * WW + ix1] * w11;
                t.e[i] = (_Float16)v;
            }
            *(uint4*)&slds[p * XS + cb * 8] = t.q;
        }
    }
    __syncthreads();

    const _Float16* arow = slds + (wave * 16 + (lane & 15)) * XS;

    v8f acc[4] = {{}, {}, {}, {}};
#pragma unroll
    for (int kb = 0; kb < 2; ++kb) {
        v16h a = load_frag(arow, kb, lane);
#pragma unroll
        for (int nt = 0; nt < 4; ++nt) {
            const _Float16* brow = wlds + ((lane & 15) + nt * 16) * NS;
            v16h bm = load_frag(brow, kb, lane);
            acc[nt] = __builtin_amdgcn_wmma_f32_16x16x32_f16(
                false, a, false, bm, (short)0, acc[nt], false, false);
        }
    }
    {
        const int pb = wave * 16 + ((lane >> 4) << 3);   // pixel base
#pragma unroll
        for (int nt = 0; nt < 4; ++nt) {
            V8F d; d.v = acc[nt];
            const int o = (lane & 15) + nt * 16;
            const float bias = end_b[o];
            float* op = out + (((size_t)b * CC + o) * SH + Y) * SW + (X0 + pb);
#pragma unroll
            for (int r = 0; r < 8; ++r)
                op[r] = d.e[r] + bias;
        }
    }
}

// ---------------------------------------------------------------------------
extern "C" void kernel_launch(void* const* d_in, const int* in_sizes, int n_in,
                              void* d_out, int out_size, void* d_ws, size_t ws_size,
                              hipStream_t stream) {
    const float* x        = (const float*)d_in[0];
    const float* offset_w = (const float*)d_in[1];
    const float* offset_b = (const float*)d_in[2];
    const float* scope_w  = (const float*)d_in[3];
    const float* end_w    = (const float*)d_in[4];
    const float* end_b    = (const float*)d_in[5];
    float* gridbuf = (float*)d_ws;   // B*4*SH*SW*2 floats = ~18.9 MB
    float* out     = (float*)d_out;

    const int nblkA = BB * HH * (WW / 16);   // 9216
    dysample_offset_kernel<<<nblkA, 128, 0, stream>>>(
        x, offset_w, offset_b, scope_w, gridbuf);

    const int nblkB = BB * SH * (SW / 64);   // 9216
    dysample_sample_gemm_kernel<<<nblkB, 128, 0, stream>>>(
        x, gridbuf, end_w, end_b, out);
}
